// AttentionMechanismBatch_42090679501302
// MI455X (gfx1250) — matrix operand order repaired and checked by reading, not compiled, not run on hardware
//
#include <hip/hip_runtime.h>

typedef _Float16 f16;
typedef __attribute__((ext_vector_type(16))) _Float16 v16h;
typedef __attribute__((ext_vector_type(8)))  _Float16 v8h;
typedef __attribute__((ext_vector_type(8)))  float    v8f;
typedef __attribute__((ext_vector_type(4)))  unsigned int u32x4;
typedef __attribute__((ext_vector_type(8)))  int i32x8;
typedef __attribute__((ext_vector_type(4)))  int i32x4;

#define B_   32
#define T_   1024
#define DIN_ 400
#define K_   10
#define L_   64
#define C_   73
#define KPAD 416      // 13 * 32
#define NKC  13
#define CPAD 80
#define ROWS 128      // rows (b*T+t) per block

#if __has_builtin(__builtin_amdgcn_tensor_load_to_lds) && __has_builtin(__builtin_amdgcn_s_wait_tensorcnt)
#define HAVE_TDM 1
#else
#define HAVE_TDM 0
#endif

// LDS layout (byte offsets), phase-aliased:
//  phase1: Wt f16[32][416] @0 (26624) | Abuf0 f32[128][32] @26624 | Abuf1 @43008 | bias f32[32] @59392
//  phase2: P  f32[128][32] @0 (16384) | PHI f16[128][64] @16384   | MT f16[80][64] @32768 (ends 43008)
#define SM_WT    0
#define SM_ABUF  26624
#define SM_ABSZ  16384
#define SM_BIAS  59392
#define SM_SIZE  59520
#define SM_P     0
#define SM_PHI   16384
#define SM_MT    32768

#if HAVE_TDM
// Issue one TDM 2D tile load: global f32 tile [128 rows x 32 cols], row stride DIN_,
// tensor_dim0 = validK so hardware zero-fills the K padding via OOB reads.
__device__ __forceinline__ void tdm_load_tile(const float* gsrc, unsigned ldsOff, int validK) {
    unsigned long long ga = (unsigned long long)(uintptr_t)gsrc;
    u32x4 g0;
    g0[0] = 1u;                                                  // count=1 (valid), no gather
    g0[1] = ldsOff;                                              // lds_addr (bytes)
    g0[2] = (unsigned)(ga & 0xffffffffu);                        // global_addr[31:0]
    g0[3] = (unsigned)((ga >> 32) & 0x1ffffffu) | (2u << 30);    // global_addr[56:32] | type=2
    i32x8 g1;
    g1[0] = (int)(2u << 16);                                     // workgroup_mask=0, data_size=4B
    g1[1] = (int)(((unsigned)validK & 0xffffu) << 16);           // tensor_dim0[15:0]
    g1[2] = (int)(((unsigned)validK >> 16) | (128u << 16));      // tensor_dim0[31:16] | tensor_dim1[15:0]
    g1[3] = (int)(32u << 16);                                    // tensor_dim1[31:16]=0 | tile_dim0=32
    g1[4] = 128;                                                 // tile_dim1=128, tile_dim2=0
    g1[5] = DIN_;                                                // tensor_dim0_stride[31:0] = 400
    g1[6] = 0;                                                   // stride0 hi | dim1_stride lo
    g1[7] = 0;
    i32x4 z4 = {0, 0, 0, 0};
#if __clang_major__ >= 23
    i32x8 z8 = {0, 0, 0, 0, 0, 0, 0, 0};
    __builtin_amdgcn_tensor_load_to_lds(g0, g1, z4, z4, z8, 0);
#else
    __builtin_amdgcn_tensor_load_to_lds(g0, g1, z4, z4, 0);
#endif
}
#endif

__global__ __launch_bounds__(256) void graves_attn_kernel(
    const float* __restrict__ X, const float* __restrict__ prevk,
    const float* __restrict__ onehot, const int* __restrict__ slen,
    const float* __restrict__ W, const float* __restrict__ bvec,
    float* __restrict__ outW, float* __restrict__ outKappa)
{
    __shared__ alignas(32) char smem[SM_SIZE];
    f16*   Wt    = (f16*)(smem + SM_WT);        // [32][KPAD]
    float* biasS = (float*)(smem + SM_BIAS);    // [32]
    float* P     = (float*)(smem + SM_P);       // [128][32]
    f16*   PHI   = (f16*)(smem + SM_PHI);       // [128][64]
    f16*   MT    = (f16*)(smem + SM_MT);        // [80][64]

    const int tid  = threadIdx.x;
    const int lane = tid & 31;
    const int wave = tid >> 5;
    const int rowbase = blockIdx.x * ROWS;      // flat row index b*T + t
    const int b = rowbase / T_;                 // whole block lies in one batch

    const int m    = lane & 15;
    const int hi16 = lane >> 4;

    // ---- stage W^T into LDS as f16, zero-padded to [32][416] ----
    for (int idx = tid; idx < 32 * KPAD; idx += 256) {
        int n = idx / KPAD, k = idx - n * KPAD;
        float v = (n < 30 && k < DIN_) ? W[k * 30 + n] : 0.f;
        Wt[idx] = (f16)v;
    }
    if (tid < 32) biasS[tid] = (tid < 30) ? bvec[tid] : 0.f;

    // ---- GEMM1: P(pre) = X[128x400] * W[400x32pad], f16 WMMA, f32 acc ----
    v8f acc0 = {}, acc1 = {};
    const int kb  = hi16 * 16;     // A-frag K base within chunk
#if HAVE_TDM
    const unsigned ldsbase = (unsigned)(uintptr_t)(void*)smem;
    if (wave == 0)
        tdm_load_tile(X + (size_t)rowbase * DIN_, ldsbase + SM_ABUF, DIN_);
#endif
    for (int kc = 0; kc < NKC; ++kc) {
        const int k0 = kc * 32;
        float* Abuf = (float*)(smem + SM_ABUF + (kc & 1) * SM_ABSZ);
#if HAVE_TDM
        if (wave == 0) {
            if (kc + 1 < NKC) {
                tdm_load_tile(X + (size_t)rowbase * DIN_ + (kc + 1) * 32,
                              ldsbase + SM_ABUF + ((kc + 1) & 1) * SM_ABSZ,
                              DIN_ - (kc + 1) * 32);
                __builtin_amdgcn_s_wait_tensorcnt(1);   // previous tile landed
            } else {
                __builtin_amdgcn_s_wait_tensorcnt(0);
            }
        }
        __syncthreads();                                // tile kc visible to all waves
#else
        __syncthreads();
        for (int i = 0; i < 16; ++i) {                  // fallback: manual staging (f32)
            int idx = tid + i * 256;
            int r = idx >> 5, c = idx & 31;
            int k = k0 + c;
            Abuf[idx] = (k < DIN_) ? X[(size_t)(rowbase + r) * DIN_ + k] : 0.f;
        }
        __syncthreads();
#endif
        // A fragment: lane m rows, K = kb..kb+15 contiguous, cvt f32->f16 here
        const float* ap = &Abuf[(wave * 16 + m) * 32 + kb];
        v8f alo = *(const v8f*)ap;
        v8f ahi = *(const v8f*)(ap + 8);
        v16h a;
        #pragma unroll
        for (int i = 0; i < 8; ++i) { a[i] = (f16)alo[i]; a[i + 8] = (f16)ahi[i]; }
        // B fragments: 16 contiguous f16 K values starting at hi16*16
        const int kbb = k0 + hi16 * 16;
        v16h b0 = *(const v16h*)&Wt[(size_t)m * KPAD + kbb];
        v16h b1 = *(const v16h*)&Wt[(size_t)(m + 16) * KPAD + kbb];
        acc0 = __builtin_amdgcn_wmma_f32_16x16x32_f16(false, a, false, b0, (short)0, acc0, false, false);
        acc1 = __builtin_amdgcn_wmma_f32_16x16x32_f16(false, a, false, b1, (short)0, acc1, false, false);
        __syncthreads();                                // all reads done before buffer reuse
    }

    // ---- bias + softplus, write P[128][32] f32 (aliases dead Wt region) ----
    {
        const int rrow = wave * 16 + 8 * hi16;
        const int c0 = m, c1 = m + 16;
        #pragma unroll
        for (int r = 0; r < 8; ++r) {
            float v0 = acc0[r] + biasS[c0];
            float v1 = acc1[r] + biasS[c1];
            v0 = fmaxf(v0, 0.f) + __logf(1.f + __expf(-fabsf(v0)));
            v1 = fmaxf(v1, 0.f) + __logf(1.f + __expf(-fabsf(v1)));
            P[(rrow + r) * 32 + c0] = v0;
            P[(rrow + r) * 32 + c1] = v1;
        }
    }
    __syncthreads();

    // ---- build masked one-hot MT[c][l] (f16), c padded to 80 ----
    {
        const int len = slen[b];
        for (int idx = tid; idx < CPAD * L_; idx += 256) {
            int c = idx >> 6, l = idx & 63;
            float v = (c < C_ && l < len) ? onehot[((size_t)b * L_ + l) * C_ + c] : 0.f;
            MT[idx] = (f16)v;
        }
    }

    // ---- per-row alpha/beta/kappa, kappa out, phi[row][64] -> LDS f16 ----
    {
        const int row  = tid >> 1;       // 0..127
        const int half = tid & 1;        // each half does 32 of the 64 l values
        const int grow = rowbase + row;
        float alpha[K_], rbeta[K_], kap[K_];
        #pragma unroll
        for (int k = 0; k < K_; ++k) {
            alpha[k] = P[row * 32 + k];
            float be = fmaxf(P[row * 32 + 10 + k], 0.01f);
            rbeta[k] = __builtin_amdgcn_rcpf(be);
            kap[k] = prevk[(size_t)grow * K_ + k] + P[row * 32 + 20 + k] * 0.04f;
        }
        if (half == 0) {
            #pragma unroll
            for (int k = 0; k < K_; ++k) outKappa[(size_t)grow * K_ + k] = kap[k];
        }
        const int l0 = half * 32;
        for (int l = l0; l < l0 + 32; ++l) {
            float u = (float)l, s = 0.f;
            #pragma unroll
            for (int k = 0; k < K_; ++k) {
                float d = kap[k] - u;
                s += alpha[k] * __expf(-d * d * rbeta[k]);
            }
            PHI[row * 64 + l] = (f16)s;
        }
    }
    __syncthreads();

    // ---- GEMM2: w[128x80] = PHI[128x64] * MT^T, 5 n-tiles x 2 k-chunks ----
    v8f wacc[5] = {};
    #pragma unroll
    for (int kc2 = 0; kc2 < 2; ++kc2) {
        const int kb2 = hi16 * 16 + kc2 * 32;
        v8h alo = *(const v8h*)&PHI[(wave * 16 + m) * 64 + kb2];
        v8h ahi = *(const v8h*)&PHI[(wave * 16 + m) * 64 + kb2 + 8];
        v16h a;
        #pragma unroll
        for (int i = 0; i < 8; ++i) { a[i] = alo[i]; a[i + 8] = ahi[i]; }
        const int kbb = hi16 * 16 + kc2 * 32;
        #pragma unroll
        for (int nt = 0; nt < 5; ++nt) {
            v16h bb = *(const v16h*)&MT[(nt * 16 + m) * 64 + kbb];
            wacc[nt] = __builtin_amdgcn_wmma_f32_16x16x32_f16(false, a, false, bb, (short)0, wacc[nt], false, false);
        }
    }
    // ---- store w (drop padded cols 73..79) ----
    {
        const int rrow = wave * 16 + 8 * hi16;
        #pragma unroll
        for (int nt = 0; nt < 5; ++nt) {
            const int col = nt * 16 + m;
            if (col < C_) {
                #pragma unroll
                for (int r = 0; r < 8; ++r)
                    outW[(size_t)(rowbase + rrow + r) * C_ + col] = wacc[nt][r];
            }
        }
    }
}

extern "C" void kernel_launch(void* const* d_in, const int* in_sizes, int n_in,
                              void* d_out, int out_size, void* d_ws, size_t ws_size,
                              hipStream_t stream) {
    const float* X      = (const float*)d_in[0];
    const float* prevk  = (const float*)d_in[1];
    const float* onehot = (const float*)d_in[2];
    const int*   slen   = (const int*)  d_in[3];
    const float* W      = (const float*)d_in[4];
    const float* bvec   = (const float*)d_in[5];
    float* outW     = (float*)d_out;
    float* outKappa = outW + (size_t)B_ * T_ * C_;
    dim3 grid(B_ * T_ / ROWS), block(256);
    hipLaunchKernelGGL(graves_attn_kernel, grid, block, 0, stream,
                       X, prevk, onehot, slen, W, bvec, outW, outKappa);
}
